// IntraModalContrastive_53575422050511
// MI455X (gfx1250) — matrix operand order repaired and checked by reading, not compiled, not run on hardware
//
#include <hip/hip_runtime.h>
#include <stdint.h>

#define B_ROWS   8192
#define HID      1024
#define PROJ     512
#define KHARD    4096
#define TEMP_INV 10.0f     // 1 / 0.1
#define NOISE_TH 9.0f      // THRESH / TEMP = 0.9 / 0.1
#define CHUNK    1024
#define LDS_PAD  8         // bf16 elements of row padding (keeps 16B align, kills bank conflicts)
#define LDS_LDK  (32 + LDS_PAD)

typedef __attribute__((ext_vector_type(16))) __bf16 v16bf;
typedef __attribute__((ext_vector_type(8)))  float  v8f;

// ---------------------------------------------------------------------------
// fp32 -> bf16 elementwise convert
// ---------------------------------------------------------------------------
__global__ __launch_bounds__(256)
void cvt_f32_bf16_kernel(const float* __restrict__ in, __bf16* __restrict__ out, int n) {
    int i = blockIdx.x * 256 + threadIdx.x;
    if (i < n) out[i] = (__bf16)in[i];
}

// ---------------------------------------------------------------------------
// Tiled transpose + convert: out[N][K] (bf16) = in[K][N] (f32)
// 32x32 tiles via LDS; K, N multiples of 32.
// ---------------------------------------------------------------------------
__global__ __launch_bounds__(256)
void transpose_cvt_kernel(const float* __restrict__ in, __bf16* __restrict__ out,
                          int Kd, int Nd) {
    __shared__ __bf16 tile[32][33];
    const int bn = blockIdx.x * 32;           // N base
    const int bk = blockIdx.y * 32;           // K base
    const int tx = threadIdx.x & 31;
    const int ty = threadIdx.x >> 5;          // 0..7
    for (int r = ty; r < 32; r += 8)
        tile[r][tx] = (__bf16)in[(size_t)(bk + r) * Nd + bn + tx];
    __syncthreads();
    for (int r = ty; r < 32; r += 8)
        out[(size_t)(bn + r) * Kd + bk + tx] = tile[tx][r];
}

// ---------------------------------------------------------------------------
// Tiled bf16 WMMA GEMM:  C[M,N] = act( A[M,K] @ Bt^T + bias ) * scale
//   A  : [M,K] row-major bf16
//   Bt : [N,K] row-major bf16  (operand B pre-transposed / naturally N-major)
// Block: 256 threads (8 waves), tile BM=128 BN=128 BK=32.
// Wave grid 2(m) x 4(n); each wave owns a 64x32 patch = 4x2 16x16 acc tiles.
// Tiles are staged with GLOBAL_LOAD_ASYNC_TO_LDS_B128 (ASYNCcnt-tracked DMA,
// no VGPR round-trip); each thread owns 4 fixed 16B transfers per K-step.
// Fragment reads are contiguous runs -> ds_load_b128.
// M,N % 128 == 0, K % 32 == 0 (true for all shapes here).
// ---------------------------------------------------------------------------
__global__ __launch_bounds__(256)
void gemm_bf16_wmma_kernel(const __bf16* __restrict__ A,
                           const __bf16* __restrict__ Bt,
                           const float*  __restrict__ bias,
                           float*  __restrict__ Cf,
                           __bf16* __restrict__ Cb,
                           int M, int N, int K,
                           int doRelu, float scale, int outBf16) {
    __shared__ __align__(16) __bf16 As[128 * LDS_LDK];
    __shared__ __align__(16) __bf16 Bs[128 * LDS_LDK];

    const int tid    = threadIdx.x;
    const int lane   = tid & 31;
    const int wave   = tid >> 5;              // 0..7
    const int wm     = (wave & 1) * 64;       // wave M offset within block tile
    const int wn     = (wave >> 1) * 32;      // wave N offset within block tile
    const int bm     = blockIdx.y * 128;
    const int bn     = blockIdx.x * 128;
    const int l      = lane & 15;
    const int hiHalf = lane >> 4;

    // staging ownership: thread copies 16B segments of rows sr and sr+64
    const int sr = tid >> 2;                  // 0..63
    const int sc = (tid & 3) * 8;             // 0,8,16,24 (bf16 units)
    const __bf16* aG0 = A  + (size_t)(bm + sr) * K + sc;
    const __bf16* aG1 = A  + (size_t)(bm + sr + 64) * K + sc;
    const __bf16* bG0 = Bt + (size_t)(bn + sr) * K + sc;
    const __bf16* bG1 = Bt + (size_t)(bn + sr + 64) * K + sc;
    // per-lane LDS byte addresses (generic LDS pointer's low 32 bits = LDS offset)
    const unsigned aL0 = (unsigned)(uintptr_t)&As[sr * LDS_LDK + sc];
    const unsigned aL1 = (unsigned)(uintptr_t)&As[(sr + 64) * LDS_LDK + sc];
    const unsigned bL0 = (unsigned)(uintptr_t)&Bs[sr * LDS_LDK + sc];
    const unsigned bL1 = (unsigned)(uintptr_t)&Bs[(sr + 64) * LDS_LDK + sc];

    v8f acc[4][2];
#pragma unroll
    for (int tm = 0; tm < 4; ++tm)
#pragma unroll
        for (int tn = 0; tn < 2; ++tn)
            acc[tm][tn] = (v8f){};

    for (int k0 = 0; k0 < K; k0 += 32) {
        // async DMA: global -> LDS, 4 x 16B per thread, tracked by ASYNCcnt
        asm volatile(
            "global_load_async_to_lds_b128 %0, %4, off\n\t"
            "global_load_async_to_lds_b128 %1, %5, off\n\t"
            "global_load_async_to_lds_b128 %2, %6, off\n\t"
            "global_load_async_to_lds_b128 %3, %7, off"
            :
            : "v"(aL0), "v"(aL1), "v"(bL0), "v"(bL1),
              "v"(aG0 + k0), "v"(aG1 + k0), "v"(bG0 + k0), "v"(bG1 + k0)
            : "memory");
        // prefetch the next K tile into L2 while the DMA runs
        if (k0 + 32 < K) {
            __builtin_prefetch(aG0 + k0 + 32, 0, 1);
            __builtin_prefetch(bG0 + k0 + 32, 0, 1);
        }
        asm volatile("s_wait_asynccnt 0x0" ::: "memory");
        __syncthreads();

        // ---- B fragments: lane (l, hiHalf) needs K = hiHalf*16 + 0..15 for N = wn+tn*16+l
        v16bf bfrag[2];
#pragma unroll
        for (int tn = 0; tn < 2; ++tn)
#pragma unroll
            for (int i = 0; i < 16; ++i)
                bfrag[tn][i] = Bs[(wn + tn * 16 + l) * LDS_LDK + hiHalf * 16 + i];

        // ---- A fragments: lane needs K = hiHalf*16 + 0..15 for M = wm+tm*16+l
#pragma unroll
        for (int tm = 0; tm < 4; ++tm) {
            v16bf af;
#pragma unroll
            for (int i = 0; i < 16; ++i) {
                int kk = hiHalf * 16 + i;
                af[i] = As[(wm + tm * 16 + l) * LDS_LDK + kk];
            }
#pragma unroll
            for (int tn = 0; tn < 2; ++tn)
                acc[tm][tn] = __builtin_amdgcn_wmma_f32_16x16x32_bf16(
                    false, af, false, bfrag[tn], (short)0, acc[tm][tn],
                    false, false);
        }
        __syncthreads();
    }

    // ---- epilogue; C layout: VGPR r, lanes0-15 -> M=r, lanes16-31 -> M=8+r ----
#pragma unroll
    for (int tm = 0; tm < 4; ++tm)
#pragma unroll
        for (int tn = 0; tn < 2; ++tn)
#pragma unroll
            for (int r = 0; r < 8; ++r) {
                int row = bm + wm + tm * 16 + hiHalf * 8 + r;
                int col = bn + wn + tn * 16 + l;
                float v = acc[tm][tn][r];
                if (bias) v += bias[col];
                if (doRelu) v = fmaxf(v, 0.0f);
                v *= scale;
                if (outBf16) Cb[(size_t)row * N + col] = (__bf16)v;
                else         Cf[(size_t)row * N + col] = v;
            }
}

// ---------------------------------------------------------------------------
// Row L2-normalize (PROJ=512 wide), fp32 in -> bf16 out (for sim GEMM)
// ---------------------------------------------------------------------------
__global__ __launch_bounds__(256)
void normalize_rows_kernel(const float* __restrict__ in, __bf16* __restrict__ out) {
    __shared__ float red[256];
    const int row = blockIdx.x;
    const float* r = in + (size_t)row * PROJ;
    float s0 = r[threadIdx.x];
    float s1 = r[threadIdx.x + 256];
    red[threadIdx.x] = s0 * s0 + s1 * s1;
    __syncthreads();
    for (int off = 128; off > 0; off >>= 1) {
        if (threadIdx.x < off) red[threadIdx.x] += red[threadIdx.x + off];
        __syncthreads();
    }
    float rn = rsqrtf(red[0]);
    out[(size_t)row * PROJ + threadIdx.x]       = (__bf16)(s0 * rn);
    out[(size_t)row * PROJ + threadIdx.x + 256] = (__bf16)(s1 * rn);
}

// ---------------------------------------------------------------------------
// Per-row hard-negative mining + weighted diagonal log-softmax CE.
// One block (1024 threads) per row. Row (8192 f32 already scaled by 1/TEMP)
// maps to monotone u32 keys in LDS; the K-th largest ranking value is found
// with a 32-step binary search over the key space (uniform control flow).
// ---------------------------------------------------------------------------
__global__ __launch_bounds__(1024)
void row_loss_kernel(const float* __restrict__ sim,    // [CHUNK][Btot]
                     const float* __restrict__ mw,     // [Btot][2]
                     int mwCol, int rowBase, int Btot, int Khard,
                     float* __restrict__ rowLoss) {
    __shared__ unsigned int keys[B_ROWS];
    __shared__ float red[1024];
    __shared__ int   cnt;
    __shared__ float sdiag;

    const int tid = threadIdx.x;
    const int row = rowBase + blockIdx.x;           // global row / label index
    const float* srow = sim + (size_t)blockIdx.x * Btot;

    // build ranking keys (diag & noise masked to 0 == -inf)
    for (int j = tid; j < Btot; j += 1024) {
        float s = srow[j];
        bool diag  = (j == row);
        bool noise = (!diag) && (s > NOISE_TH);
        unsigned int k = 0u;
        if (!(diag || noise)) {
            unsigned int u = __float_as_uint(s);
            k = (u & 0x80000000u) ? ~u : (u | 0x80000000u);
        }
        keys[j] = k;
    }
    __syncthreads();

    // binary search: largest T with count(key >= T) >= Khard
    unsigned long long lo = 0ull, hi = 0xFFFFFFFFull;
    for (int it = 0; it < 33 && lo < hi; ++it) {
        unsigned long long mid = lo + ((hi - lo + 1ull) >> 1);
        if (tid == 0) cnt = 0;
        __syncthreads();
        unsigned int T = (unsigned int)mid;
        int c = 0;
        for (int j = tid; j < Btot; j += 1024) c += (keys[j] >= T) ? 1 : 0;
        atomicAdd(&cnt, c);
        __syncthreads();
        int total = cnt;
        __syncthreads();
        if (total >= Khard) lo = mid; else hi = mid - 1ull;
    }
    const unsigned int T = (unsigned int)lo;

    // pass 1: weights + row max (and stash diagonal logit)
    float lmax = -INFINITY;
    for (int j = tid; j < Btot; j += 1024) {
        float s = srow[j];
        float w;
        if (j == row)            w = 1.0f;
        else if (s > NOISE_TH)   w = 0.5f;
        else if (T != 0u && keys[j] >= T) w = 1.5f;
        else                     w = 1.0f;
        float wsv = s * w;
        if (j == row) sdiag = wsv;
        lmax = fmaxf(lmax, wsv);
    }
    red[tid] = lmax;
    __syncthreads();
    for (int off = 512; off > 0; off >>= 1) {
        if (tid < off) red[tid] = fmaxf(red[tid], red[tid + off]);
        __syncthreads();
    }
    const float m = red[0];
    __syncthreads();

    // pass 2: sum of exp
    float lsum = 0.0f;
    for (int j = tid; j < Btot; j += 1024) {
        float s = srow[j];
        float w;
        if (j == row)            w = 1.0f;
        else if (s > NOISE_TH)   w = 0.5f;
        else if (T != 0u && keys[j] >= T) w = 1.5f;
        else                     w = 1.0f;
        lsum += __expf(s * w - m);
    }
    red[tid] = lsum;
    __syncthreads();
    for (int off = 512; off > 0; off >>= 1) {
        if (tid < off) red[tid] += red[tid + off];
        __syncthreads();
    }
    if (tid == 0) {
        float ce = -(sdiag - m - logf(red[0]));
        rowLoss[row] = ce * mw[(size_t)row * 2 + mwCol];
    }
}

// ---------------------------------------------------------------------------
// Deterministic final reduction: (mean_t + mean_v) / 2
// ---------------------------------------------------------------------------
__global__ __launch_bounds__(256)
void finalize_kernel(const float* __restrict__ rlT, const float* __restrict__ rlV,
                     float* __restrict__ out, int Bt) {
    __shared__ float red[256];
    float a = 0.0f;
    for (int j = threadIdx.x; j < Bt; j += 256) a += rlT[j] + rlV[j];
    red[threadIdx.x] = a;
    __syncthreads();
    for (int off = 128; off > 0; off >>= 1) {
        if (threadIdx.x < off) red[threadIdx.x] += red[threadIdx.x + off];
        __syncthreads();
    }
    if (threadIdx.x == 0) out[0] = red[0] / (2.0f * (float)Bt);
}

// ---------------------------------------------------------------------------
// Host orchestration
// ---------------------------------------------------------------------------
static inline int cdiv_i(int a, int b) { return (a + b - 1) / b; }

extern "C" void kernel_launch(void* const* d_in, const int* in_sizes, int n_in,
                              void* d_out, int out_size, void* d_ws, size_t ws_size,
                              hipStream_t stream) {
    (void)in_sizes; (void)n_in; (void)out_size; (void)ws_size;

    const float* id_emb  = (const float*)d_in[0];
    const float* text_f  = (const float*)d_in[1];
    const float* vis_f   = (const float*)d_in[2];
    const float* mw      = (const float*)d_in[3];
    const float* Wa_t = (const float*)d_in[4];  const float* ba_t = (const float*)d_in[5];
    const float* Wa_v = (const float*)d_in[6];  const float* ba_v = (const float*)d_in[7];
    const float* W1_t = (const float*)d_in[8];  const float* b1_t = (const float*)d_in[9];
    const float* W2_t = (const float*)d_in[10]; const float* b2_t = (const float*)d_in[11];
    const float* W1_v = (const float*)d_in[12]; const float* b1_v = (const float*)d_in[13];
    const float* W2_v = (const float*)d_in[14]; const float* b2_v = (const float*)d_in[15];

    // workspace carving (256B aligned)
    char* ws = (char*)d_ws;
    size_t off = 0;
    auto carve = [&](size_t bytes) -> void* {
        void* p = ws + off;
        off += (bytes + 255) & ~(size_t)255;
        return p;
    };
    __bf16* id_bf   = (__bf16*)carve((size_t)B_ROWS * HID * 2);
    __bf16* feat_bf = (__bf16*)carve((size_t)B_ROWS * 2048 * 2);
    __bf16* WaT_bf  = (__bf16*)carve((size_t)HID * 2048 * 2);   // [HID][F]
    __bf16* W1T_bf  = (__bf16*)carve((size_t)HID * HID * 2);    // [HID][HID]
    __bf16* W2T_bf  = (__bf16*)carve((size_t)PROJ * HID * 2);   // [PROJ][HID]
    __bf16* x_bf    = (__bf16*)carve((size_t)B_ROWS * HID * 2);
    __bf16* h_bf    = (__bf16*)carve((size_t)B_ROWS * HID * 2);
    float*  proj_f  = (float*) carve((size_t)B_ROWS * PROJ * 4);
    __bf16* mp_bf   = (__bf16*)carve((size_t)B_ROWS * PROJ * 2);
    __bf16* ip_bf   = (__bf16*)carve((size_t)B_ROWS * PROJ * 2);
    float*  simbuf  = (float*) carve((size_t)CHUNK * B_ROWS * 4);
    float*  rl_t    = (float*) carve((size_t)B_ROWS * 4);
    float*  rl_v    = (float*) carve((size_t)B_ROWS * 4);

    const dim3 blk256(256);

    // id embeddings -> bf16 once
    cvt_f32_bf16_kernel<<<cdiv_i(B_ROWS * HID, 256), blk256, 0, stream>>>(
        id_emb, id_bf, B_ROWS * HID);

    auto run_modal = [&](const float* feat, int F,
                         const float* Wa, const float* ba,
                         const float* W1, const float* b1,
                         const float* W2, const float* b2,
                         int mwCol, float* rl) {
        cvt_f32_bf16_kernel<<<cdiv_i(B_ROWS * F, 256), blk256, 0, stream>>>(feat, feat_bf, B_ROWS * F);
        // weights converted + transposed once: Bt layouts [N][K]
        transpose_cvt_kernel<<<dim3(HID / 32, F / 32),   blk256, 0, stream>>>(Wa, WaT_bf, F,   HID);
        transpose_cvt_kernel<<<dim3(HID / 32, HID / 32), blk256, 0, stream>>>(W1, W1T_bf, HID, HID);
        transpose_cvt_kernel<<<dim3(PROJ / 32, HID / 32),blk256, 0, stream>>>(W2, W2T_bf, HID, PROJ);

        // adapter: modal_feat = feat @ Wa + ba          (bf16 out)
        gemm_bf16_wmma_kernel<<<dim3(HID / 128, B_ROWS / 128), blk256, 0, stream>>>(
            feat_bf, WaT_bf, ba, nullptr, x_bf, B_ROWS, HID, F, 0, 1.0f, 1);
        // projector L1: h = relu(modal_feat @ W1 + b1)  (bf16 out)
        gemm_bf16_wmma_kernel<<<dim3(HID / 128, B_ROWS / 128), blk256, 0, stream>>>(
            x_bf, W1T_bf, b1, nullptr, h_bf, B_ROWS, HID, HID, 1, 1.0f, 1);
        // projector L2: proj = h @ W2 + b2              (f32 out)
        gemm_bf16_wmma_kernel<<<dim3(PROJ / 128, B_ROWS / 128), blk256, 0, stream>>>(
            h_bf, W2T_bf, b2, proj_f, nullptr, B_ROWS, PROJ, HID, 0, 1.0f, 0);
        normalize_rows_kernel<<<B_ROWS, blk256, 0, stream>>>(proj_f, mp_bf);

        // id path through the same projector
        gemm_bf16_wmma_kernel<<<dim3(HID / 128, B_ROWS / 128), blk256, 0, stream>>>(
            id_bf, W1T_bf, b1, nullptr, h_bf, B_ROWS, HID, HID, 1, 1.0f, 1);
        gemm_bf16_wmma_kernel<<<dim3(PROJ / 128, B_ROWS / 128), blk256, 0, stream>>>(
            h_bf, W2T_bf, b2, proj_f, nullptr, B_ROWS, PROJ, HID, 0, 1.0f, 0);
        normalize_rows_kernel<<<B_ROWS, blk256, 0, stream>>>(proj_f, ip_bf);

        // sim = (id_proj @ modal_proj^T) / TEMP, chunked by 1024 rows, fused loss
        // (mp_bf is [8192][512] = naturally [N][K] for this GEMM)
        for (int c = 0; c < B_ROWS / CHUNK; ++c) {
            gemm_bf16_wmma_kernel<<<dim3(B_ROWS / 128, CHUNK / 128), blk256, 0, stream>>>(
                ip_bf + (size_t)c * CHUNK * PROJ, mp_bf, nullptr,
                simbuf, nullptr, CHUNK, B_ROWS, PROJ, 0, TEMP_INV, 0);
            row_loss_kernel<<<CHUNK, 1024, 0, stream>>>(
                simbuf, mw, mwCol, c * CHUNK, B_ROWS, KHARD, rl);
        }
    };

    run_modal(text_f, 768,  Wa_t, ba_t, W1_t, b1_t, W2_t, b2_t, 0, rl_t);
    run_modal(vis_f,  2048, Wa_v, ba_v, W1_v, b1_v, W2_v, b2_v, 1, rl_v);

    finalize_kernel<<<1, blk256, 0, stream>>>(rl_t, rl_v, (float*)d_out, B_ROWS);
}
